// LinearSumAssignment_16209206575564
// MI455X (gfx1250) — matrix operand order repaired and checked by reading, not compiled, not run on hardware
//
#include <hip/hip_runtime.h>

#define NF    162      // H*W
#define NP    163      // n+1 (virtual col 0)
#define CCH   2048
#define BATCH 64
#define PHALF 81
#define INFC  1000000000.0f

typedef __attribute__((ext_vector_type(16))) _Float16 v16h;
typedef __attribute__((ext_vector_type(8)))  float    v8f;

union FragH { v16h h; int4 q[2]; };

// ---------------------------------------------------------------------------
// Kernel 1: per-column channel norms + rank-based binary top-half mask.
// Normalization in compute_mask is monotone, so bMask depends only on ranks.
// ---------------------------------------------------------------------------
__global__ void k_norm_mask(const float* __restrict__ feat,
                            float* __restrict__ qn, float* __restrict__ bm) {
  __shared__ float norms[NF];
  const int b = blockIdx.x;
  const int t = threadIdx.x;
  if (t < NF) {
    const float* p = feat + (size_t)b * CCH * NF + t;
    float s0 = 0.f, s1 = 0.f, s2 = 0.f, s3 = 0.f;
    for (int c = 0; c < CCH; c += 4) {
      float x0 = p[(size_t)(c + 0) * NF];
      float x1 = p[(size_t)(c + 1) * NF];
      float x2 = p[(size_t)(c + 2) * NF];
      float x3 = p[(size_t)(c + 3) * NF];
      s0 += x0 * x0; s1 += x1 * x1; s2 += x2 * x2; s3 += x3 * x3;
    }
    norms[t] = sqrtf(s0 + s1 + s2 + s3);
  }
  __syncthreads();
  if (t < NF) {
    float v = norms[t];
    int cnt = 0;
    for (int k = 0; k < NF; ++k) cnt += (norms[k] < v) ? 1 : 0;
    qn[b * NF + t] = v;
    bm[b * NF + t] = (cnt >= PHALF) ? 1.0f : 0.0f;
  }
}

// ---------------------------------------------------------------------------
// Kernel 2: sim[b,i,j] = bMask_i * <feat[b,:,i], feat[pos_b,:,j]> / (|.||.|)
// One wave per (b, 16x16 tile). f32 accuracy via hi/lo f16 split: 3 WMMAs
// (Ahi*Bhi + Ahi*Blo + Alo*Bhi) per K=32 chunk, f32 accumulate.
// ---------------------------------------------------------------------------
__global__ void k_sim_wmma(const float* __restrict__ feat,
                           const int* __restrict__ pos_ind,
                           const float* __restrict__ qn,
                           const float* __restrict__ bm,
                           float* __restrict__ sim) {
  __shared__ alignas(16) _Float16 sAhi[16 * 32];
  __shared__ alignas(16) _Float16 sAlo[16 * 32];
  __shared__ alignas(16) _Float16 sBhi[16 * 32];
  __shared__ alignas(16) _Float16 sBlo[16 * 32];

  const int tj = blockIdx.x, ti = blockIdx.y, b = blockIdx.z;
  const int pb = pos_ind[b];
  const int lane = threadIdx.x;
  const int col  = lane & 15;
  const int half = lane >> 4;

  // clamp padded columns (tiles cover 176 >= 162); padded outputs never stored
  const int ca = min(ti * 16 + col, NF - 1);
  const int cb = min(tj * 16 + col, NF - 1);
  const float* baseA = feat + (size_t)b  * CCH * NF;
  const float* baseB = feat + (size_t)pb * CCH * NF;

  v8f acc = {};
  for (int kc = 0; kc < CCH; kc += 32) {
    union { _Float16 h[16]; int4 q[2]; } ahi, alo, bhi, blo;
#pragma unroll
    for (int kk = 0; kk < 16; ++kk) {
      const int c = kc + half * 16 + kk;
      const float xa = baseA[(size_t)c * NF + ca];
      const float xb = baseB[(size_t)c * NF + cb];
      const _Float16 ha = (_Float16)xa;
      const _Float16 hb = (_Float16)xb;
      ahi.h[kk] = ha; alo.h[kk] = (_Float16)(xa - (float)ha);
      bhi.h[kk] = hb; blo.h[kk] = (_Float16)(xb - (float)hb);
    }
    __syncthreads();
    // LDS layout: s[col][k], 32 f16 (64B) per column; lane writes 16 f16
    const int st = col * 32 + half * 16;
    *(int4*)&sAhi[st] = ahi.q[0]; *(int4*)&sAhi[st + 8] = ahi.q[1];
    *(int4*)&sAlo[st] = alo.q[0]; *(int4*)&sAlo[st + 8] = alo.q[1];
    *(int4*)&sBhi[st] = bhi.q[0]; *(int4*)&sBhi[st + 8] = bhi.q[1];
    *(int4*)&sBlo[st] = blo.q[0]; *(int4*)&sBlo[st + 8] = blo.q[1];
    __syncthreads();

    // A frag (16x32 f16): M=lane%16; K runs [8*half, +8) and [16+8*half, +8)
    FragH fAhi, fAlo, fBhi, fBlo;
    const int a0 = col * 32 + 8 * half, a1 = a0 + 16;
    fAhi.q[0] = *(const int4*)&sAhi[a0]; fAhi.q[1] = *(const int4*)&sAhi[a1];
    fAlo.q[0] = *(const int4*)&sAlo[a0]; fAlo.q[1] = *(const int4*)&sAlo[a1];
    // B frag (32x16 f16): N=lane%16; K runs [8*half, +8) and [16+8*half, +8)
    const int b0 = col * 32 + 8 * half, b1 = b0 + 16;
    fBhi.q[0] = *(const int4*)&sBhi[b0]; fBhi.q[1] = *(const int4*)&sBhi[b1];
    fBlo.q[0] = *(const int4*)&sBlo[b0]; fBlo.q[1] = *(const int4*)&sBlo[b1];

    acc = __builtin_amdgcn_wmma_f32_16x16x32_f16(false, fAhi.h, false, fBhi.h,
                                                 (short)0, acc, false, false);
    acc = __builtin_amdgcn_wmma_f32_16x16x32_f16(false, fAhi.h, false, fBlo.h,
                                                 (short)0, acc, false, false);
    acc = __builtin_amdgcn_wmma_f32_16x16x32_f16(false, fAlo.h, false, fBhi.h,
                                                 (short)0, acc, false, false);
    __syncthreads();
  }

  // Epilogue: C/D layout: VGPR r -> M = r + 8*half, N = lane%16
  const int j = tj * 16 + col;
  if (j < NF) {
    const float qj = fmaxf(qn[pb * NF + j], 1e-12f);
#pragma unroll
    for (int r = 0; r < 8; ++r) {
      const int i = ti * 16 + r + 8 * half;
      if (i < NF) {
        const float qi = fmaxf(qn[b * NF + i], 1e-12f);
        sim[((size_t)b * NF + i) * NF + j] = bm[b * NF + i] * (acc[r] / (qi * qj));
      }
    }
  }
}

// ---------------------------------------------------------------------------
// Kernel 3: Jonker-Volgenant assignment (e-maxx formulation), one wave per
// batch. Each lane owns 6 columns (j = lane + 32*s). v/minv/used register-
// resident; u/p/way in LDS; argmin via shfl_xor butterfly with jnp.argmin
// tie-breaking (first index).
// ---------------------------------------------------------------------------
__global__ void k_hungarian(const float* __restrict__ sim, float* __restrict__ pd) {
  const int b = blockIdx.x;
  const int lane = threadIdx.x;
  __shared__ float u[NP];
  __shared__ int   p[NP];
  __shared__ int   way[NP];

  for (int s = 0; s < 6; ++s) {
    const int j = lane + 32 * s;
    if (j < NP) { u[j] = 0.0f; p[j] = 0; }
  }
  float v[6];
#pragma unroll
  for (int s = 0; s < 6; ++s) v[s] = 0.0f;
  __syncthreads();

  const float* simb = sim + (size_t)b * NF * NF;

  for (int i = 1; i <= NF; ++i) {
    if (lane == 0) p[0] = i;
    float minv[6];
#pragma unroll
    for (int s = 0; s < 6; ++s) minv[s] = INFC;
    unsigned usedMask = 0;
    for (int s = 0; s < 6; ++s) {
      const int j = lane + 32 * s;
      if (j < NP) way[j] = 0;
    }
    __syncthreads();

    int j0 = 0;
    bool done = false;
    while (!done) {
      if ((j0 & 31) == lane) usedMask |= 1u << (j0 >> 5);  // used[j0] = true
      const int i0 = p[j0];
      const float ui0 = u[i0];
      // cur = costp[i0] - u[i0] - v ; cost = -sim
#pragma unroll
      for (int s = 0; s < 6; ++s) {
        const int j = lane + 32 * s;
        const bool valid = (j >= 1) && (j <= NF) && !((usedMask >> s) & 1u);
        if (valid) {
          const float cur = -simb[(size_t)(i0 - 1) * NF + (j - 1)] - ui0 - v[s];
          if (cur < minv[s]) { minv[s] = cur; way[j] = j0; }
        }
      }
      // argmin over where(valid, minv, INF), first-index tie-break
      float bestv = INFC; int bestj = NP;
#pragma unroll
      for (int s = 0; s < 6; ++s) {
        const int j = lane + 32 * s;
        const bool valid = (j >= 1) && (j <= NF) && !((usedMask >> s) & 1u);
        const float mv = valid ? minv[s] : INFC;
        if (mv < bestv || (mv == bestv && j < bestj)) { bestv = mv; bestj = j; }
      }
      for (int off = 16; off; off >>= 1) {
        const float ov = __shfl_xor(bestv, off, 32);
        const int   oj = __shfl_xor(bestj, off, 32);
        if (ov < bestv || (ov == bestv && oj < bestj)) { bestv = ov; bestj = oj; }
      }
      const float delta = bestv;
      const int j1 = bestj;
      // used j: u[p[j]] += delta (distinct rows), v[j] -= delta
      // unused j: minv[j] -= delta
#pragma unroll
      for (int s = 0; s < 6; ++s) {
        const int j = lane + 32 * s;
        if (j < NP) {
          if ((usedMask >> s) & 1u) {
            v[s] -= delta;
            const int r = p[j];
            u[r] += delta;
          } else {
            minv[s] -= delta;
          }
        }
      }
      __syncthreads();
      j0 = j1;
      done = (p[j1] == 0);
    }
    // augment along way[] chain
    __syncthreads();
    if (lane == 0) {
      int jj = j0;
      while (jj != 0) {
        const int j1w = way[jj];
        p[jj] = p[j1w];
        jj = j1w;
      }
    }
    __syncthreads();
  }

  // assigned.sum = sum_j sim[p[j]-1, j-1];  pos_dis = 1 - sum/81
  float partial = 0.0f;
  for (int s = 0; s < 6; ++s) {
    const int j = lane + 32 * s;
    if (j >= 1 && j <= NF) {
      const int r = p[j];
      partial += simb[(size_t)(r - 1) * NF + (j - 1)];
    }
  }
  for (int off = 16; off; off >>= 1) partial += __shfl_xor(partial, off, 32);
  if (lane == 0) pd[b] = 1.0f - partial / (float)PHALF;
}

// ---------------------------------------------------------------------------
// Kernel 4: mean over 64 per-batch values -> scalar output
// ---------------------------------------------------------------------------
__global__ void k_final(const float* __restrict__ pd, float* __restrict__ out) {
  const int lane = threadIdx.x;
  float s = pd[lane] + pd[lane + 32];
  for (int off = 16; off; off >>= 1) s += __shfl_xor(s, off, 32);
  if (lane == 0) out[0] = s / (float)BATCH;
}

// ---------------------------------------------------------------------------
extern "C" void kernel_launch(void* const* d_in, const int* in_sizes, int n_in,
                              void* d_out, int out_size, void* d_ws, size_t ws_size,
                              hipStream_t stream) {
  (void)in_sizes; (void)n_in; (void)out_size; (void)ws_size;
  const float* feat = (const float*)d_in[0];
  const int*   pos  = (const int*)d_in[1];
  // d_in[2] (neg_ind) unused by the reference output

  float* ws  = (float*)d_ws;
  float* sim = ws;                                   // 64*162*162
  float* qn  = ws + (size_t)BATCH * NF * NF;         // 64*162
  float* bm  = qn + (size_t)BATCH * NF;              // 64*162
  float* pd  = bm + (size_t)BATCH * NF;              // 64

  k_norm_mask<<<dim3(BATCH), dim3(192), 0, stream>>>(feat, qn, bm);
  k_sim_wmma<<<dim3(11, 11, BATCH), dim3(32), 0, stream>>>(feat, pos, qn, bm, sim);
  k_hungarian<<<dim3(BATCH), dim3(32), 0, stream>>>(sim, pd);
  k_final<<<dim3(1), dim3(32), 0, stream>>>(pd, (float*)d_out);
}
